// AdaptiveCantorModalityFusion_46059229283217
// MI455X (gfx1250) — matrix operand order repaired and checked by reading, not compiled, not run on hardware
//
#include <hip/hip_runtime.h>
#include <hip/hip_bf16.h>
#include <math.h>

// ---------------------------------------------------------------------------
// Types
// ---------------------------------------------------------------------------
typedef __bf16 bf16;
typedef __attribute__((ext_vector_type(8)))  bf16         v8bf;
typedef __attribute__((ext_vector_type(16))) bf16         v16bf;
typedef __attribute__((ext_vector_type(8)))  float        v8f;
typedef __attribute__((ext_vector_type(4)))  unsigned int u32x4;
typedef __attribute__((ext_vector_type(4)))  int          i32x4;

// Async global->LDS staging (CDNA5), if the toolchain exposes the builtin.
#if defined(__has_builtin)
#if __has_builtin(__builtin_amdgcn_global_load_async_to_lds_b128)
#define ATHENA_ASYNC_LDS 1
#endif
#endif
#ifndef ATHENA_ASYNC_LDS
#define ATHENA_ASYNC_LDS 0
#endif

// ---------------------------------------------------------------------------
// Helpers
// ---------------------------------------------------------------------------
__device__ __forceinline__ bf16 f2b(float f) {
  unsigned u = __builtin_bit_cast(unsigned, f);
  unsigned r = u + 0x7FFFu + ((u >> 16) & 1u);   // round-to-nearest-even
  return __builtin_bit_cast(bf16, (unsigned short)(r >> 16));
}
__device__ __forceinline__ float b2f(bf16 b) {
  unsigned u = ((unsigned)__builtin_bit_cast(unsigned short, b)) << 16;
  return __builtin_bit_cast(float, u);
}
__device__ __forceinline__ v16bf cat16(v8bf lo, v8bf hi) {
  v16bf r;
#pragma unroll
  for (int i = 0; i < 8; ++i) { r[i] = lo[i]; r[i + 8] = hi[i]; }
  return r;
}

// ---------------------------------------------------------------------------
// f32 -> bf16 elementwise conversion (activations)
// ---------------------------------------------------------------------------
__global__ __launch_bounds__(256) void f32_to_bf16_kernel(
    const float* __restrict__ in, bf16* __restrict__ out, long long n) {
  long long i = (long long)blockIdx.x * blockDim.x + threadIdx.x;
  const long long str = (long long)gridDim.x * blockDim.x;
  for (; i < n; i += str) out[i] = f2b(in[i]);
}

// ---------------------------------------------------------------------------
// f32 [K][N] -> bf16 [N][K] transpose-convert (weights, one-time).
// 32x32 tiles through LDS; K, N multiples of 32.
// ---------------------------------------------------------------------------
__global__ __launch_bounds__(256) void transpose_f32_bf16_kernel(
    const float* __restrict__ in, bf16* __restrict__ out, int K, int N) {
  __shared__ float t[32][33];
  const int k0 = blockIdx.y * 32;
  const int n0 = blockIdx.x * 32;
  const int tid = threadIdx.x;
#pragma unroll
  for (int j = 0; j < 4; ++j) {
    const int e = tid + 256 * j;
    const int kk = e >> 5, nn = e & 31;
    t[kk][nn] = in[(long long)(k0 + kk) * N + n0 + nn];
  }
  __syncthreads();
#pragma unroll
  for (int j = 0; j < 4; ++j) {
    const int e = tid + 256 * j;
    const int nn = e >> 5, kk = e & 31;
    out[(long long)(n0 + nn) * K + k0 + kk] = f2b(t[kk][nn]);
  }
}

// ---------------------------------------------------------------------------
// Generic bf16 WMMA GEMM:  C[M,N] = act( A[M,K] @ B[K,N] + bias )
//   B supplied PRE-TRANSPOSED as Bt[N][K] (row length = K).
//   A rows mapped two-level: row r -> (b = r / rpb, s = r % rpb)
//     A addr = A + b*abs_ + s*lda ;  C addr = Cx + b*cbs_ + s*ldc
//   Requires: N % 128 == 0, K % 64 == 0.  M guarded.
// ---------------------------------------------------------------------------
struct GemmP {
  const bf16* A; const bf16* Bt; const float* bias;
  float* Cf; bf16* Cb;
  int M, N, K;
  int rpb, lda, ldc, act;       // act: 0 = none, 1 = exact GELU
  long long abs_, cbs_;
};

#define TM 128
#define TN 128
#define TK 64

#if ATHENA_ASYNC_LDS
#define AS1 __attribute__((address_space(1)))
#define AS3 __attribute__((address_space(3)))
__device__ __forceinline__ void wait_async0() {
#if __has_builtin(__builtin_amdgcn_s_wait_asynccnt)
  __builtin_amdgcn_s_wait_asynccnt(0);
#else
  asm volatile("s_wait_asynccnt 0x0" ::: "memory");
#endif
}
// Issue 8 x GLOBAL_LOAD_ASYNC_TO_LDS_B128 per thread (ASYNCcnt-tracked):
// data flows cache -> LDS without a VGPR round-trip.  Out-of-range A rows are
// clamped: garbage only feeds C rows the epilogue never stores.
__device__ __forceinline__ void stage_async(const GemmP& p, int tileM, int tileN,
    int kt, int ar, int ac, bf16* Asl, bf16* Bsl) {
  const int k0 = kt * TK;
  int R = tileM + ar;
  if (R >= p.M) R = p.M - 1;
  const int bb = R / p.rpb;
  const int ss = R - bb * p.rpb;
  const bf16* ap = p.A + (long long)bb * p.abs_ + (long long)ss * p.lda + k0 + ac;
  const bf16* bp = p.Bt + (long long)(tileN + ar) * p.K + k0 + ac;
  bf16* al = &Asl[ar * TK + ac];
  bf16* bl = &Bsl[ar * TK + ac];
#pragma unroll
  for (int j = 0; j < 4; ++j) {
    __builtin_amdgcn_global_load_async_to_lds_b128(
        (AS1 i32x4*)(ap + 8 * j), (AS3 i32x4*)(al + 8 * j), 0, 0);
    __builtin_amdgcn_global_load_async_to_lds_b128(
        (AS1 i32x4*)(bp + 8 * j), (AS3 i32x4*)(bl + 8 * j), 0, 0);
  }
}
#else
__device__ __forceinline__ void stage_load(const GemmP& p, int tileM, int tileN,
    int kt, int ar, int ac, u32x4 av[4], u32x4 bv[4]) {
  const int k0 = kt * TK;
  const int R = tileM + ar;
  if (R < p.M) {
    const int bb = R / p.rpb;
    const int ss = R - bb * p.rpb;
    const bf16* ap = p.A + (long long)bb * p.abs_ + (long long)ss * p.lda + k0 + ac;
#pragma unroll
    for (int j = 0; j < 4; ++j) av[j] = *(const u32x4*)(ap + 8 * j);
  } else {
#pragma unroll
    for (int j = 0; j < 4; ++j) { av[j][0] = av[j][1] = av[j][2] = av[j][3] = 0u; }
  }
  const bf16* bp = p.Bt + (long long)(tileN + ar) * p.K + k0 + ac;
#pragma unroll
  for (int j = 0; j < 4; ++j) bv[j] = *(const u32x4*)(bp + 8 * j);
}
__device__ __forceinline__ void stage_store(bf16* As, bf16* Bs, int ar, int ac,
    const u32x4 av[4], const u32x4 bv[4]) {
#pragma unroll
  for (int j = 0; j < 4; ++j) {
    *(u32x4*)&As[ar * TK + ac + 8 * j] = av[j];
    *(u32x4*)&Bs[ar * TK + ac + 8 * j] = bv[j];
  }
}
#endif

__global__ __launch_bounds__(256) void gemm_bf16_wmma(GemmP p) {
  __shared__ bf16 As[2][TM * TK];        // [m][k], 16 KB each buffer
  __shared__ bf16 Bs[2][TN * TK];        // [n][k], 16 KB each buffer

  const int tid  = threadIdx.x;
  const int lane = tid & 31;
  const int wid  = tid >> 5;
  const int hl   = lane >> 4;            // wave half (0/1)
  const int l16  = lane & 15;
  const int wm   = wid >> 2;             // 0..1  -> 64-row strip
  const int wn   = wid & 3;              // 0..3  -> 32-col strip
  const int tileM = blockIdx.y * TM;
  const int tileN = blockIdx.x * TN;
  const int nk = p.K / TK;

  // staging: 2 threads per row, 32 contiguous bf16 (4 x b128) each
  const int ar = tid >> 1;
  const int ac = (tid & 1) * 32;

  v8f acc[4][2];
#pragma unroll
  for (int mt = 0; mt < 4; ++mt)
#pragma unroll
    for (int nt = 0; nt < 2; ++nt)
#pragma unroll
      for (int e = 0; e < 8; ++e) acc[mt][nt][e] = 0.0f;

#if ATHENA_ASYNC_LDS
  stage_async(p, tileM, tileN, 0, ar, ac, As[0], Bs[0]);
  wait_async0();
  __syncthreads();
#else
  u32x4 av[4], bv[4];
  stage_load(p, tileM, tileN, 0, ar, ac, av, bv);
  stage_store(As[0], Bs[0], ar, ac, av, bv);
  __syncthreads();
#endif

  for (int kt = 0; kt < nk; ++kt) {
    const int cur = kt & 1;
    const bool more = (kt + 1) < nk;
#if ATHENA_ASYNC_LDS
    if (more)  // async-copy next K-tile into the other buffer while computing
      stage_async(p, tileM, tileN, kt + 1, ar, ac, As[cur ^ 1], Bs[cur ^ 1]);
#else
    if (more)  // prefetch next K-tile into registers while computing
      stage_load(p, tileM, tileN, kt + 1, ar, ac, av, bv);
#endif

    const bf16* Al = As[cur];
    const bf16* Bl = Bs[cur];

#pragma unroll
    for (int ks = 0; ks < 2; ++ks) {      // two 16x16x32 K-slices per tile
      const int ko = ks * 32;
      // B fragments (32x16): lane l16 = column; lanes 0-15 K{0..7,16..23}, 16-31 K{8..15,24..31}
      v16bf bfrag[2];
#pragma unroll
      for (int nt = 0; nt < 2; ++nt) {
        const int n = wn * 32 + nt * 16 + l16;
        v8bf lo = *(const v8bf*)&Bl[n * TK + ko + hl * 8];
        v8bf hi = *(const v8bf*)&Bl[n * TK + ko + 16 + hl * 8];
        bfrag[nt] = cat16(lo, hi);
      }
      // A fragments (16x32): lanes 0-15 K{0..7,16..23}, 16-31 K{8..15,24..31}
#pragma unroll
      for (int mt = 0; mt < 4; ++mt) {
        const int m = wm * 64 + mt * 16 + l16;
        v8bf lo = *(const v8bf*)&Al[m * TK + ko + hl * 8];
        v8bf hi = *(const v8bf*)&Al[m * TK + ko + 16 + hl * 8];
        v16bf afrag = cat16(lo, hi);
        acc[mt][0] = __builtin_amdgcn_wmma_f32_16x16x32_bf16(
            false, afrag, false, bfrag[0], (short)0, acc[mt][0], false, false);
        acc[mt][1] = __builtin_amdgcn_wmma_f32_16x16x32_bf16(
            false, afrag, false, bfrag[1], (short)0, acc[mt][1], false, false);
      }
    }

#if ATHENA_ASYNC_LDS
    if (more) wait_async0();
#else
    if (more) stage_store(As[cur ^ 1], Bs[cur ^ 1], ar, ac, av, bv);
#endif
    __syncthreads();
  }

  // Epilogue.  C/D layout: m = v + 8*half, n = lane&15
#pragma unroll
  for (int mt = 0; mt < 4; ++mt) {
#pragma unroll
    for (int nt = 0; nt < 2; ++nt) {
      const int gn = tileN + wn * 32 + nt * 16 + l16;
#pragma unroll
      for (int v = 0; v < 8; ++v) {
        const int gm = tileM + wm * 64 + mt * 16 + hl * 8 + v;
        if (gm < p.M) {
          float x = acc[mt][nt][v];
          if (p.bias) x += p.bias[gn];
          if (p.act == 1) x = 0.5f * x * (1.0f + erff(x * 0.70710678118654752f));
          const int bb = gm / p.rpb;
          const int ss = gm - bb * p.rpb;
          const long long o = (long long)bb * p.cbs_ + (long long)ss * p.ldc + gn;
          if (p.Cf) p.Cf[o] = x;
          if (p.Cb) p.Cb[o] = f2b(x);
        }
      }
    }
  }
}

// ---------------------------------------------------------------------------
// Gate reduce: gate[r] = sigmoid( h[r,:256] . w2 + b2 ).  One wave32 per row.
// ---------------------------------------------------------------------------
__global__ __launch_bounds__(256) void gate_reduce_kernel(
    const bf16* __restrict__ h, const float* __restrict__ w2,
    const float* __restrict__ b2, float* __restrict__ gate, int R) {
  const int wid = threadIdx.x >> 5, lane = threadIdx.x & 31;
  const int row = blockIdx.x * 8 + wid;
  if (row >= R) return;
  float s = 0.0f;
#pragma unroll
  for (int j = 0; j < 8; ++j) {
    const int c = lane + j * 32;
    s += b2f(h[(long long)row * 256 + c]) * w2[c];
  }
  s += __shfl_xor(s, 16); s += __shfl_xor(s, 8); s += __shfl_xor(s, 4);
  s += __shfl_xor(s, 2);  s += __shfl_xor(s, 1);
  if (lane == 0) gate[row] = 1.0f / (1.0f + expf(-(s + b2[0])));
}

// ---------------------------------------------------------------------------
// Modulate + modality embedding + pad to S=256, pack into stacked bf16 [B,M,S,D]
// ---------------------------------------------------------------------------
__global__ __launch_bounds__(256) void modulate_pack_kernel(
    const bf16* __restrict__ p_all, const float* __restrict__ gate_all,
    const float* __restrict__ alphas, const float* __restrict__ emb,
    bf16* __restrict__ stacked) {
  const long long N = 16LL * 4 * 256 * 1024;
  long long e = (long long)blockIdx.x * blockDim.x + threadIdx.x;
  const long long str = (long long)gridDim.x * blockDim.x;
  for (; e < N; e += str) {
    const int d = (int)(e & 1023);
    const long long t = e >> 10;
    const int s = (int)(t & 255);
    const long long t2 = t >> 8;
    const int m = (int)(t2 & 3);
    const int b = (int)(t2 >> 2);
    const int seq = (m < 2) ? 77 : 256;
    float outv = 0.0f;
    if (s < seq) {
      const int rows_before = (m < 2) ? (m * 1232) : (2464 + (m - 2) * 4096);
      const int row = rows_before + b * seq + s;
      const float pv = b2f(p_all[(long long)row * 1024 + d]);
      const float g  = gate_all[row];
      const float a  = 1.0f / (1.0f + expf(-alphas[m]));
      outv = pv * (g * a + (1.0f - a)) + emb[m * 1024 + d];
    }
    stacked[e] = f2b(outv);
  }
}

// ---------------------------------------------------------------------------
// Per-position 4x4 cross-modality attention.  One wave32 per (b, s, h).
// Lanes cover head_dim 64 as d = lane and lane+32.
// ---------------------------------------------------------------------------
__global__ __launch_bounds__(256) void attn_kernel(
    const bf16* __restrict__ Q, const bf16* __restrict__ K,
    const bf16* __restrict__ V, const float* __restrict__ temp,
    const float* __restrict__ betas, bf16* __restrict__ merged) {
  const int wid = threadIdx.x >> 5, lane = threadIdx.x & 31;
  const int idx = blockIdx.x * 8 + wid;           // 0 .. 65535 = B*S*H
  const int h = idx & 15;
  const int s = (idx >> 4) & 255;
  const int b = idx >> 12;
  const int col = h * 64 + lane;

  float q[4][2], k[4][2], v[4][2];
#pragma unroll
  for (int m = 0; m < 4; ++m) {
    const long long r = ((long long)(b * 4 + m) * 256 + s) * 1024 + col;
    q[m][0] = b2f(Q[r]); q[m][1] = b2f(Q[r + 32]);
    k[m][0] = b2f(K[r]); k[m][1] = b2f(K[r + 32]);
    v[m][0] = b2f(V[r]); v[m][1] = b2f(V[r + 32]);
  }
  const float scale = 1.0f / (8.0f * fabsf(temp[0]));   // sqrt(64)=8
  float sc[4][4];
#pragma unroll
  for (int m = 0; m < 4; ++m) {
#pragma unroll
    for (int w = 0; w < 4; ++w) {
      float pp = q[m][0] * k[w][0] + q[m][1] * k[w][1];
      pp += __shfl_xor(pp, 16); pp += __shfl_xor(pp, 8);
      pp += __shfl_xor(pp, 4);  pp += __shfl_xor(pp, 2);
      pp += __shfl_xor(pp, 1);
      float bias = (((m ^ w) & 1) != 0) ? -1e9f : 0.0f;  // binding parity mask
      if (m == 0 && w == 2) bias += betas[0];
      if (m == 1 && w == 3) bias += betas[1];
      sc[m][w] = pp * scale + bias;
    }
  }
#pragma unroll
  for (int m = 0; m < 4; ++m) {
    const float mx = fmaxf(fmaxf(sc[m][0], sc[m][1]), fmaxf(sc[m][2], sc[m][3]));
    const float e0 = expf(sc[m][0] - mx), e1 = expf(sc[m][1] - mx);
    const float e2 = expf(sc[m][2] - mx), e3 = expf(sc[m][3] - mx);
    const float inv = 1.0f / (e0 + e1 + e2 + e3);
    const float c0 = (e0 * v[0][0] + e1 * v[1][0] + e2 * v[2][0] + e3 * v[3][0]) * inv;
    const float c1 = (e0 * v[0][1] + e1 * v[1][1] + e2 * v[2][1] + e3 * v[3][1]) * inv;
    const long long r = ((long long)(b * 4 + m) * 256 + s) * 1024 + col;
    merged[r] = f2b(c0); merged[r + 32] = f2b(c1);
  }
}

// ---------------------------------------------------------------------------
// Host orchestration
// ---------------------------------------------------------------------------
extern "C" void kernel_launch(void* const* d_in, const int* in_sizes, int n_in,
                              void* d_out, int out_size, void* d_ws, size_t ws_size,
                              hipStream_t stream) {
  (void)in_sizes; (void)n_in; (void)out_size; (void)ws_size;

  const float* x[4]  = {(const float*)d_in[0], (const float*)d_in[3],
                        (const float*)d_in[6], (const float*)d_in[9]};
  const float* Wp[4] = {(const float*)d_in[1], (const float*)d_in[4],
                        (const float*)d_in[7], (const float*)d_in[10]};
  const float* bp[4] = {(const float*)d_in[2], (const float*)d_in[5],
                        (const float*)d_in[8], (const float*)d_in[11]};
  const float* Wg1    = (const float*)d_in[12];
  const float* bg1    = (const float*)d_in[13];
  const float* Wg2    = (const float*)d_in[14];
  const float* bg2    = (const float*)d_in[15];
  const float* alphas = (const float*)d_in[16];
  const float* emb    = (const float*)d_in[17];
  const float* Wq = (const float*)d_in[18]; const float* bq = (const float*)d_in[19];
  const float* Wk = (const float*)d_in[20]; const float* bk = (const float*)d_in[21];
  const float* Wv = (const float*)d_in[22]; const float* bv = (const float*)d_in[23];
  const float* temp  = (const float*)d_in[24];
  const float* betas = (const float*)d_in[25];
  const float* Wout  = (const float*)d_in[26];
  const float* bout  = (const float*)d_in[27];
  float* out = (float*)d_out;

  const int SEQ[4]  = {77, 77, 256, 256};
  const int DIN[4]  = {768, 1280, 2048, 2048};
  const int ROWS[4] = {16 * 77, 16 * 77, 16 * 256, 16 * 256};
  const long long PROW[4] = {0, 1232, 2464, 6560};   // row offsets in packed p/h/gate
  const int COFF[4] = {0, 77, 154, 410};             // seq offsets in output (666)

  char* wsb = (char*)d_ws;
  size_t off = 0;
  auto alloc = [&](size_t bytes) -> void* {
    off = (off + 255) & ~(size_t)255;
    void* pp = wsb + off;
    off += bytes;
    return pp;
  };

  bf16* xb[4];  bf16* Wpt[4];
  for (int i = 0; i < 4; ++i) xb[i]  = (bf16*)alloc((size_t)ROWS[i] * DIN[i] * 2);
  for (int i = 0; i < 4; ++i) Wpt[i] = (bf16*)alloc((size_t)DIN[i] * 1024 * 2);
  bf16*  Wg1t  = (bf16*)alloc(4LL * 1024 * 256 * 2);   // [4][256][1024]
  bf16*  Wqt   = (bf16*)alloc(1024LL * 1024 * 2);
  bf16*  Wkt   = (bf16*)alloc(1024LL * 1024 * 2);
  bf16*  Wvt   = (bf16*)alloc(1024LL * 1024 * 2);
  bf16*  Woutt = (bf16*)alloc(4LL * 1024 * 1024 * 2);  // [4][1024][1024] (transposed)
  bf16*  p_b   = (bf16*)alloc(10656LL * 1024 * 2);
  bf16*  h_b   = (bf16*)alloc(10656LL * 256 * 2);
  float* gate  = (float*)alloc(10656LL * 4);
  bf16*  stk   = (bf16*)alloc(16LL * 4 * 256 * 1024 * 2);
  bf16*  Qb    = (bf16*)alloc(16777216LL * 2);
  bf16*  Kb    = (bf16*)alloc(16777216LL * 2);
  bf16*  Vb    = (bf16*)alloc(16777216LL * 2);
  bf16*  mrg   = (bf16*)alloc(16777216LL * 2);

  auto conv = [&](const float* src, bf16* dst, long long n) {
    int blocks = (int)((n + 255) / 256);
    if (blocks > 16384) blocks = 16384;
    f32_to_bf16_kernel<<<blocks, 256, 0, stream>>>(src, dst, n);
  };
  auto transp = [&](const float* src, bf16* dst, int K, int N) {
    transpose_f32_bf16_kernel<<<dim3(N / 32, K / 32), dim3(256), 0, stream>>>(
        src, dst, K, N);
  };
  auto gemm = [&](const bf16* A, const bf16* Bt, const float* bias,
                  float* Cf, bf16* Cb, int M, int N, int Kd, int rpb,
                  long long abs_, int lda, long long cbs_, int ldc, int act) {
    GemmP gp;
    gp.A = A; gp.Bt = Bt; gp.bias = bias; gp.Cf = Cf; gp.Cb = Cb;
    gp.M = M; gp.N = N; gp.K = Kd;
    gp.rpb = rpb; gp.lda = lda; gp.ldc = ldc; gp.act = act;
    gp.abs_ = abs_; gp.cbs_ = cbs_;
    gemm_bf16_wmma<<<dim3(N / 128, (M + 127) / 128), dim3(256), 0, stream>>>(gp);
  };

  // 1) activations -> bf16; weights -> transposed bf16 (one-time)
  for (int i = 0; i < 4; ++i) conv(x[i], xb[i], (long long)ROWS[i] * DIN[i]);
  for (int i = 0; i < 4; ++i) transp(Wp[i], Wpt[i], DIN[i], 1024);
  for (int i = 0; i < 4; ++i)
    transp(Wg1 + (long long)i * 1024 * 256, Wg1t + (long long)i * 256 * 1024, 1024, 256);
  transp(Wq, Wqt, 1024, 1024);
  transp(Wk, Wkt, 1024, 1024);
  transp(Wv, Wvt, 1024, 1024);
  for (int i = 0; i < 4; ++i)
    transp(Wout + (long long)i * 1024 * 1024, Woutt + (long long)i * 1024 * 1024,
           1024, 1024);

  // 2) input projections p_i = x_i @ Wp_i + bp_i   (bf16 out)
  for (int i = 0; i < 4; ++i)
    gemm(xb[i], Wpt[i], bp[i], nullptr, p_b + PROW[i] * 1024,
         ROWS[i], 1024, DIN[i], ROWS[i], 0, DIN[i], 0, 1024, 0);

  // 3) gate MLP layer 1: h = GELU(p @ Wg1[i] + bg1[i])
  for (int i = 0; i < 4; ++i)
    gemm(p_b + PROW[i] * 1024, Wg1t + (long long)i * 256 * 1024, bg1 + i * 256,
         nullptr, h_b + PROW[i] * 256,
         ROWS[i], 256, 1024, ROWS[i], 0, 1024, 0, 256, 1);

  // 4) gate = sigmoid(h @ Wg2[i] + bg2[i])
  for (int i = 0; i < 4; ++i)
    gate_reduce_kernel<<<(ROWS[i] + 7) / 8, 256, 0, stream>>>(
        h_b + PROW[i] * 256, Wg2 + i * 256, bg2 + i, gate + PROW[i], ROWS[i]);

  // 5) modulate + embed + pad -> stacked bf16 [B,M,S,D]
  modulate_pack_kernel<<<65536, 256, 0, stream>>>(p_b, gate, alphas, emb, stk);

  // 6) Q/K/V projections over all 16384 rows
  gemm(stk, Wqt, bq, nullptr, Qb, 16384, 1024, 1024, 16384, 0, 1024, 0, 1024, 0);
  gemm(stk, Wkt, bk, nullptr, Kb, 16384, 1024, 1024, 16384, 0, 1024, 0, 1024, 0);
  gemm(stk, Wvt, bv, nullptr, Vb, 16384, 1024, 1024, 16384, 0, 1024, 0, 1024, 0);

  // 7) per-position cross-modality attention -> merged bf16 [B,M,S,D]
  attn_kernel<<<8192, 256, 0, stream>>>(Qb, Kb, Vb, temp, betas, mrg);

  // 8) output projections, sliced rows scattered into [B, 666, D] f32
  for (int i = 0; i < 4; ++i)
    gemm(mrg + (long long)i * 256 * 1024, Woutt + (long long)i * 1024 * 1024,
         bout + i * 1024,
         out + (long long)COFF[i] * 1024, nullptr,
         ROWS[i], 1024, 1024, SEQ[i],
         4LL * 256 * 1024 /* A batch stride */, 1024,
         666LL * 1024    /* C batch stride */, 1024, 0);
}
